// MoEMLP_21672404975654
// MI455X (gfx1250) — matrix operand order repaired and checked by reading, not compiled, not run on hardware
//
#include <hip/hip_runtime.h>
#include <hip/hip_bf16.h>

// ---------------- problem constants (from reference) ----------------
#define T_TOK 4096      // B*S
#define DDIM  1024
#define HDIM  1024
#define NEXP  16
#define TOPK  2
#define CAP   640       // ceil(T*K/E * 1.25)
#define CBUF  1280      // K * CAP

typedef __attribute__((ext_vector_type(16))) __bf16       v16bf;
typedef __attribute__((ext_vector_type(8)))  float        v8f;
typedef __attribute__((ext_vector_type(4)))  unsigned int u32x4;
typedef int vsi4 __attribute__((vector_size(16)));   // matches builtin's V4i param

union FragBF { u32x4 u[2]; v16bf v; };

#if __has_builtin(__builtin_amdgcn_global_load_async_to_lds_b128)
#define USE_ASYNC_LDS 1
#else
#define USE_ASYNC_LDS 0
#endif

__device__ __forceinline__ unsigned short f2bf(float f) {
    unsigned int u = __float_as_uint(f);
    unsigned int r = u + 0x7FFFu + ((u >> 16) & 1u);   // round-to-nearest-even
    return (unsigned short)(r >> 16);
}

// 16-byte global->LDS copy: async (ASYNCcnt) via gfx1250 GLOBAL_LOAD_ASYNC_TO_LDS_B128.
__device__ __forceinline__ void copy16_g2l(const unsigned short* g, unsigned short* l) {
#if USE_ASYNC_LDS
    __builtin_amdgcn_global_load_async_to_lds_b128(
        (__attribute__((address_space(1))) vsi4*)(vsi4*)const_cast<unsigned short*>(g),
        (__attribute__((address_space(3))) vsi4*)(vsi4*)l,
        0, 0);
#else
    *reinterpret_cast<u32x4*>(l) = *reinterpret_cast<const u32x4*>(g);
#endif
}

__device__ __forceinline__ void stage_wait() {
#if USE_ASYNC_LDS
    asm volatile("s_wait_asynccnt 0x0" ::: "memory");
#endif
    __syncthreads();
}

// ---------------- 0) weight convert+transpose: [E][K][N] f32 -> [E][N][K] bf16 ----------------
__global__ __launch_bounds__(256) void moe_wconv(const float* __restrict__ src,
                                                 unsigned short* __restrict__ dst) {
    __shared__ __attribute__((aligned(16))) unsigned short t[64 * 72];
    const int e = blockIdx.z, k0 = blockIdx.x * 64, n0 = blockIdx.y * 64;
    const int tid = threadIdx.x;
    const float* s = src + ((size_t)e * 1024 + k0) * 1024 + n0;
#pragma unroll
    for (int i = 0; i < 16; ++i) {
        int idx = tid + i * 256;
        int kk = idx >> 6, nn = idx & 63;
        t[nn * 72 + kk] = f2bf(s[(size_t)kk * 1024 + nn]);
    }
    __syncthreads();
    unsigned short* d = dst + ((size_t)e * 1024 + n0) * 1024 + k0;
#pragma unroll
    for (int i = 0; i < 4; ++i) {
        int c = tid + i * 256;                  // 1024 chunks of 4 halves
        int nn = c >> 4, kc = (c & 15) * 4;
        unsigned long long pk = (unsigned long long)t[nn * 72 + kc]
                              | ((unsigned long long)t[nn * 72 + kc + 1] << 16)
                              | ((unsigned long long)t[nn * 72 + kc + 2] << 32)
                              | ((unsigned long long)t[nn * 72 + kc + 3] << 48);
        *reinterpret_cast<unsigned long long*>(d + (size_t)nn * 1024 + kc) = pk;
    }
}

// ---------------- 1) router: logits -> softmax -> top2 ----------------
__global__ __launch_bounds__(256) void moe_router(
        const float* __restrict__ x, const float* __restrict__ rw,
        int* __restrict__ topk_e, float* __restrict__ topk_w) {
    int wave = threadIdx.x >> 5, lane = threadIdx.x & 31;
    int t = blockIdx.x * 8 + wave;
    const float* xrow = x + (size_t)t * DDIM;
    float acc[NEXP];
#pragma unroll
    for (int e = 0; e < NEXP; ++e) acc[e] = 0.f;
    for (int d = lane; d < DDIM; d += 32) {
        float xv = xrow[d];
        const float* r = rw + (size_t)d * NEXP;
#pragma unroll
        for (int e = 0; e < NEXP; ++e) acc[e] += xv * r[e];
    }
#pragma unroll
    for (int e = 0; e < NEXP; ++e) {
#pragma unroll
        for (int off = 16; off >= 1; off >>= 1)
            acc[e] += __shfl_xor(acc[e], off, 32);
    }
    if (lane == 0) {
        float m = acc[0];
#pragma unroll
        for (int e = 1; e < NEXP; ++e) m = fmaxf(m, acc[e]);
        float p[NEXP], s = 0.f;
#pragma unroll
        for (int e = 0; e < NEXP; ++e) { p[e] = __expf(acc[e] - m); s += p[e]; }
        float inv = 1.f / s;
#pragma unroll
        for (int e = 0; e < NEXP; ++e) p[e] *= inv;
        int i0 = 0; float b0 = p[0];
#pragma unroll
        for (int e = 1; e < NEXP; ++e) if (p[e] > b0) { b0 = p[e]; i0 = e; }
        int i1 = (i0 == 0) ? 1 : 0; float b1 = p[i1];
#pragma unroll
        for (int e = 0; e < NEXP; ++e) if (e != i0 && p[e] > b1) { b1 = p[e]; i1 = e; }
        topk_e[t * 2 + 0] = i0; topk_e[t * 2 + 1] = i1;
        topk_w[t * 2 + 0] = b0; topk_w[t * 2 + 1] = b1;
    }
}

// ---------------- 2) ordered capacity assignment (single block) ----------------
__global__ void moe_assign(const int* __restrict__ topk_e, const float* __restrict__ topk_w,
                           int* __restrict__ slot_of, int* __restrict__ dispatch,
                           float* __restrict__ wbuf) {
    for (int i = threadIdx.x; i < NEXP * CBUF; i += blockDim.x) {
        dispatch[i] = T_TOK;     // padding token
        wbuf[i] = 0.f;
    }
    __syncthreads();
    if (threadIdx.x == 0) {
        int cntk[TOPK][NEXP], cnt[NEXP];
        for (int k = 0; k < TOPK; ++k)
            for (int e = 0; e < NEXP; ++e) cntk[k][e] = 0;
        for (int e = 0; e < NEXP; ++e) cnt[e] = 0;
        for (int t = 0; t < T_TOK; ++t) {
            for (int k = 0; k < TOPK; ++k) {
                int idx = t * TOPK + k;
                int e = topk_e[idx];
                bool keep = cntk[k][e] < CAP;     // per-k-channel rank, as in reference
                cntk[k][e]++;
                if (keep) {
                    int s = cnt[e]++;             // slot over kept entries (both channels)
                    dispatch[e * CBUF + s] = t;
                    wbuf[e * CBUF + s] = topk_w[idx];
                    slot_of[idx] = s;
                } else {
                    slot_of[idx] = -1;
                }
            }
        }
    }
}

// ---------------- 3) gather xe (bf16) ----------------
__global__ __launch_bounds__(256) void moe_gather(
        const float* __restrict__ x, const int* __restrict__ dispatch,
        unsigned short* __restrict__ xe) {
    int row = blockIdx.x;               // e*CBUF + c
    int tok = dispatch[row];
    int d0 = threadIdx.x * 4;
    unsigned long long pack = 0ull;
    if (tok < T_TOK) {
        const float4 v = *reinterpret_cast<const float4*>(x + (size_t)tok * DDIM + d0);
        pack = (unsigned long long)f2bf(v.x)
             | ((unsigned long long)f2bf(v.y) << 16)
             | ((unsigned long long)f2bf(v.z) << 32)
             | ((unsigned long long)f2bf(v.w) << 48);
    }
    *reinterpret_cast<unsigned long long*>(xe + (size_t)row * DDIM + d0) = pack;
}

// ---------------- 4) gate/up GEMM + SiLU (bf16 WMMA) ----------------
// weights pre-transposed bf16: gwt/uwt = [E][H][D]  (row n, contiguous k)
// block: 256 thr (8 waves), computes 64(rows of Cbuf) x 64(cols of H) of `hidden`
__global__ __launch_bounds__(256) void moe_gateup(
        const unsigned short* __restrict__ xe, const unsigned short* __restrict__ gwt,
        const unsigned short* __restrict__ uwt, unsigned short* __restrict__ hidden) {
    __shared__ __attribute__((aligned(16))) unsigned short ldsG[64 * 40]; // [n][k0..31]
    __shared__ __attribute__((aligned(16))) unsigned short ldsU[64 * 40];
    const int e = blockIdx.z, row0 = blockIdx.y * 64, col0 = blockIdx.x * 64;
    const int tid = threadIdx.x, wave = tid >> 5, lane = tid & 31;
    const int rband = wave & 3, cpair = wave >> 2;

    const unsigned short* arow =
        xe + ((size_t)e * CBUF + row0 + rband * 16 + (lane & 15)) * DDIM;
    const int ab = (lane < 16) ? 0 : 16;           // A frag K split (contiguous k = 16h + s, matches B)
    const int kb = (lane < 16) ? 0 : 16;           // B frag K split (contiguous k = 16h + s, matches A)
    const int nl0 = (cpair * 2) * 16 + (lane & 15);
    const int nl1 = nl0 + 16;

    // staging: thread -> (row srow in [0,64), 16B chunk sch in [0,4)) of a 64x32-half tile
    const int srow = tid >> 2, sch = tid & 3;
    const unsigned short* gG = gwt + ((size_t)e * HDIM + col0 + srow) * DDIM + sch * 8;
    const unsigned short* gU = uwt + ((size_t)e * HDIM + col0 + srow) * DDIM + sch * 8;
    unsigned short* lG = &ldsG[srow * 40 + sch * 8];
    unsigned short* lU = &ldsU[srow * 40 + sch * 8];

    v8f aG0 = {0,0,0,0,0,0,0,0}, aG1 = {0,0,0,0,0,0,0,0};
    v8f aU0 = {0,0,0,0,0,0,0,0}, aU1 = {0,0,0,0,0,0,0,0};

    for (int k0 = 0; k0 < DDIM; k0 += 32) {
        copy16_g2l(gG + k0, lG);
        copy16_g2l(gU + k0, lU);
        if (k0 + 32 < DDIM) {   // prefetch next K-tile (global_prefetch_b8)
            __builtin_prefetch(gG + k0 + 32, 0, 1);
            __builtin_prefetch(gU + k0 + 32, 0, 1);
        }
        stage_wait();

        FragBF fa;
        fa.u[0] = *reinterpret_cast<const u32x4*>(arow + k0 + ab);
        fa.u[1] = *reinterpret_cast<const u32x4*>(arow + k0 + ab + 8);

        FragBF b;
        const u32x4* p;
        p = reinterpret_cast<const u32x4*>(&ldsG[nl0 * 40 + kb]); b.u[0] = p[0]; b.u[1] = p[1];
        aG0 = __builtin_amdgcn_wmma_f32_16x16x32_bf16(false, fa.v, false, b.v, (short)0, aG0, false, false);
        p = reinterpret_cast<const u32x4*>(&ldsG[nl1 * 40 + kb]); b.u[0] = p[0]; b.u[1] = p[1];
        aG1 = __builtin_amdgcn_wmma_f32_16x16x32_bf16(false, fa.v, false, b.v, (short)0, aG1, false, false);
        p = reinterpret_cast<const u32x4*>(&ldsU[nl0 * 40 + kb]); b.u[0] = p[0]; b.u[1] = p[1];
        aU0 = __builtin_amdgcn_wmma_f32_16x16x32_bf16(false, fa.v, false, b.v, (short)0, aU0, false, false);
        p = reinterpret_cast<const u32x4*>(&ldsU[nl1 * 40 + kb]); b.u[0] = p[0]; b.u[1] = p[1];
        aU1 = __builtin_amdgcn_wmma_f32_16x16x32_bf16(false, fa.v, false, b.v, (short)0, aU1, false, false);
        __syncthreads();
    }

    // epilogue: hidden = silu(gate) * up, stored bf16. C layout: VGPR i -> M=i / i+8
    const int mbase = row0 + rband * 16 + ((lane < 16) ? 0 : 8);
    const int nA = col0 + cpair * 32 + (lane & 15);
#pragma unroll
    for (int i = 0; i < 8; ++i) {
        int m = mbase + i;
        size_t base = ((size_t)e * CBUF + m) * HDIM;
        float g0 = aG0[i], u0 = aU0[i];
        float h0 = (g0 / (1.f + __expf(-g0))) * u0;
        hidden[base + nA] = f2bf(h0);
        float g1 = aG1[i], u1 = aU1[i];
        float h1 = (g1 / (1.f + __expf(-g1))) * u1;
        hidden[base + nA + 16] = f2bf(h1);
    }
}

// ---------------- 5) down GEMM + row scale (bf16 WMMA) ----------------
// dwt = [E][D][H] bf16 (row n=d, contiguous k=h)
__global__ __launch_bounds__(256) void moe_down(
        const unsigned short* __restrict__ hidden, const unsigned short* __restrict__ dwt,
        const float* __restrict__ wbuf, float* __restrict__ ye) {
    __shared__ __attribute__((aligned(16))) unsigned short ldsD[64 * 40];
    const int e = blockIdx.z, row0 = blockIdx.y * 64, col0 = blockIdx.x * 64;
    const int tid = threadIdx.x, wave = tid >> 5, lane = tid & 31;
    const int rband = wave & 3, cpair = wave >> 2;

    const unsigned short* arow =
        hidden + ((size_t)e * CBUF + row0 + rband * 16 + (lane & 15)) * HDIM;
    const int ab = (lane < 16) ? 0 : 16;
    const int kb = (lane < 16) ? 0 : 16;
    const int nl0 = (cpair * 2) * 16 + (lane & 15);
    const int nl1 = nl0 + 16;

    const int srow = tid >> 2, sch = tid & 3;
    const unsigned short* gD = dwt + ((size_t)e * DDIM + col0 + srow) * HDIM + sch * 8;
    unsigned short* lD = &ldsD[srow * 40 + sch * 8];

    v8f a0 = {0,0,0,0,0,0,0,0}, a1 = {0,0,0,0,0,0,0,0};

    for (int k0 = 0; k0 < HDIM; k0 += 32) {
        copy16_g2l(gD + k0, lD);
        if (k0 + 32 < HDIM)
            __builtin_prefetch(gD + k0 + 32, 0, 1);
        stage_wait();

        FragBF fa;
        fa.u[0] = *reinterpret_cast<const u32x4*>(arow + k0 + ab);
        fa.u[1] = *reinterpret_cast<const u32x4*>(arow + k0 + ab + 8);

        FragBF b;
        const u32x4* p;
        p = reinterpret_cast<const u32x4*>(&ldsD[nl0 * 40 + kb]); b.u[0] = p[0]; b.u[1] = p[1];
        a0 = __builtin_amdgcn_wmma_f32_16x16x32_bf16(false, fa.v, false, b.v, (short)0, a0, false, false);
        p = reinterpret_cast<const u32x4*>(&ldsD[nl1 * 40 + kb]); b.u[0] = p[0]; b.u[1] = p[1];
        a1 = __builtin_amdgcn_wmma_f32_16x16x32_bf16(false, fa.v, false, b.v, (short)0, a1, false, false);
        __syncthreads();
    }

    const int mbase = row0 + rband * 16 + ((lane < 16) ? 0 : 8);
    const int nA = col0 + cpair * 32 + (lane & 15);
#pragma unroll
    for (int i = 0; i < 8; ++i) {
        int m = mbase + i;
        float wv = wbuf[e * CBUF + m];
        size_t base = ((size_t)e * CBUF + m) * DDIM;
        ye[base + nA]      = a0[i] * wv;
        ye[base + nA + 16] = a1[i] * wv;
    }
}

// ---------------- 6) deterministic combine (no atomics) ----------------
__global__ __launch_bounds__(256) void moe_combine(
        const int* __restrict__ topk_e, const int* __restrict__ slot_of,
        const float* __restrict__ ye, float* __restrict__ out) {
    int t = blockIdx.x;
    for (int d = threadIdx.x; d < DDIM; d += blockDim.x) {
        float a = 0.f;
#pragma unroll
        for (int k = 0; k < TOPK; ++k) {
            int idx = t * TOPK + k;
            int s = slot_of[idx];
            if (s >= 0) {
                int e = topk_e[idx];
                a += ye[((size_t)e * CBUF + s) * DDIM + d];
            }
        }
        out[(size_t)t * DDIM + d] = a;
    }
}

// ---------------- launcher ----------------
extern "C" void kernel_launch(void* const* d_in, const int* in_sizes, int n_in,
                              void* d_out, int out_size, void* d_ws, size_t ws_size,
                              hipStream_t stream) {
    const float* x  = (const float*)d_in[0];
    const float* rw = (const float*)d_in[1];
    const float* gw = (const float*)d_in[2];
    const float* uw = (const float*)d_in[3];
    const float* dw = (const float*)d_in[4];
    float* out = (float*)d_out;
    char* ws = (char*)d_ws;

    // workspace layout (256B aligned)
    size_t off = 0;
    auto alloc = [&](size_t bytes) { size_t o = off; off = (off + bytes + 255) & ~(size_t)255; return o; };
    int*            topk_e  = (int*)           (ws + alloc((size_t)T_TOK * TOPK * 4));
    float*          topk_w  = (float*)         (ws + alloc((size_t)T_TOK * TOPK * 4));
    int*            slot_of = (int*)           (ws + alloc((size_t)T_TOK * TOPK * 4));
    int*            dispatch= (int*)           (ws + alloc((size_t)NEXP * CBUF * 4));
    float*          wbuf    = (float*)         (ws + alloc((size_t)NEXP * CBUF * 4));
    unsigned short* xe      = (unsigned short*)(ws + alloc((size_t)NEXP * CBUF * DDIM * 2));
    unsigned short* hidden  = (unsigned short*)(ws + alloc((size_t)NEXP * CBUF * HDIM * 2));
    float*          ye      = (float*)         (ws + alloc((size_t)NEXP * CBUF * DDIM * 4));
    unsigned short* gwt     = (unsigned short*)(ws + alloc((size_t)NEXP * DDIM * HDIM * 2));
    unsigned short* uwt     = (unsigned short*)(ws + alloc((size_t)NEXP * DDIM * HDIM * 2));
    unsigned short* dwt     = (unsigned short*)(ws + alloc((size_t)NEXP * HDIM * DDIM * 2));
    (void)in_sizes; (void)n_in; (void)out_size; (void)ws_size;

    {   // one-time weight convert+transpose to bf16 column-major
        dim3 grid(1024 / 64, 1024 / 64, NEXP);
        moe_wconv<<<grid, 256, 0, stream>>>(gw, gwt);
        moe_wconv<<<grid, 256, 0, stream>>>(uw, uwt);
        moe_wconv<<<grid, 256, 0, stream>>>(dw, dwt);
    }

    moe_router <<<T_TOK / 8, 256, 0, stream>>>(x, rw, topk_e, topk_w);
    moe_assign <<<1, 256, 0, stream>>>(topk_e, topk_w, slot_of, dispatch, wbuf);
    moe_gather <<<NEXP * CBUF, 256, 0, stream>>>(x, dispatch, xe);
    {
        dim3 grid(HDIM / 64, CBUF / 64, NEXP);
        moe_gateup <<<grid, 256, 0, stream>>>(xe, gwt, uwt, hidden);
    }
    {
        dim3 grid(DDIM / 64, CBUF / 64, NEXP);
        moe_down <<<grid, 256, 0, stream>>>(hidden, dwt, wbuf, ye);
    }
    moe_combine<<<T_TOK, 256, 0, stream>>>(topk_e, slot_of, ye, out);
}
